// MultiHeadCrossAttention_50964081934885
// MI455X (gfx1250) — matrix operand order repaired and checked by reading, not compiled, not run on hardware
//
#include <hip/hip_runtime.h>
#include <hip/hip_bf16.h>

typedef __attribute__((ext_vector_type(16))) _Float16 v16h;
typedef __attribute__((ext_vector_type(8)))  _Float16 v8h;
typedef __attribute__((ext_vector_type(4)))  _Float16 v4h;
typedef __attribute__((ext_vector_type(8)))  float    v8f;
typedef __attribute__((ext_vector_type(4)))  float    v4f;
typedef __attribute__((ext_vector_type(4)))  int      v4i;

typedef __attribute__((address_space(1))) v4i as1_v4i;   // global int4
typedef __attribute__((address_space(3))) v4i as3_v4i;   // LDS int4

#if defined(__has_builtin)
#  if __has_builtin(__builtin_amdgcn_global_load_async_to_lds_b128)
#    define HAVE_ASYNC_LDS 1
#  endif
#endif
#ifndef HAVE_ASYNC_LDS
#  define HAVE_ASYNC_LDS 0
#endif

static __device__ __forceinline__ v16h combine8(v8h lo, v8h hi) {
    v16h r;
#pragma unroll
    for (int i = 0; i < 8; ++i) { r[i] = lo[i]; r[i + 8] = hi[i]; }
    return r;
}

static __device__ __forceinline__ v8f wmma_f16(v16h a, v16h b, v8f c) {
    return __builtin_amdgcn_wmma_f32_16x16x32_f16(false, a, false, b, (short)0, c, false, false);
}

// ---------------------------------------------------------------------------
// Projection GEMM: Y[M,1024] = X[M,1024] @ W[1024,1024]^T + bias
//   mode 0: store f16 into [B,H,S,64] head layout (for q/k/v)
//   mode 1: store f32 row-major [M,1024]           (for o-proj)
// XHALF selects f16 vs f32 input activations.
// ---------------------------------------------------------------------------
template <bool XHALF>
__global__ __launch_bounds__(256)
void proj_gemm(const float* __restrict__ Xf, const _Float16* __restrict__ Xh,
               const float* __restrict__ W, const float* __restrict__ bias,
               _Float16* __restrict__ outH, float* __restrict__ outF,
               int M, int S, int mode)
{
    constexpr int LD = 40;                 // half stride (pad -> 16B aligned, conflict-free)
    __shared__ _Float16 As[128 * LD];
    __shared__ _Float16 Bs[128 * LD];

    const int t    = threadIdx.x;
    const int lane = t & 31;
    const int wv   = t >> 5;
    const int ll   = lane & 15;
    const int hl   = lane >> 4;
    const int m0   = blockIdx.y * 128;
    const int n0   = blockIdx.x * 128;
    const int wm   = (wv & 3) * 32;        // 4 wave-rows of 32
    const int wn   = (wv >> 2) * 64;       // 2 wave-cols of 64

    v8f acc[2][4];
#pragma unroll
    for (int i = 0; i < 2; ++i)
#pragma unroll
        for (int n = 0; n < 4; ++n) acc[i][n] = (v8f){0.f,0.f,0.f,0.f,0.f,0.f,0.f,0.f};

    for (int kt = 0; kt < 1024 / 32; ++kt) {
        const int k0 = kt * 32;
        __syncthreads();
        // --- stage A (convert to f16) ---
#pragma unroll
        for (int j = 0; j < 4; ++j) {
            const int e   = t + j * 256;       // 1024 float4-chunks of the 128x32 tile
            const int row = e >> 3;
            const int cc  = (e & 7) * 4;
            v4h hv;
            if (XHALF) {
                const _Float16* p = Xh + (size_t)(m0 + row) * 1024 + k0 + cc;
#pragma unroll
                for (int q = 0; q < 4; ++q) hv[q] = p[q];
            } else {
                v4f x = *(const v4f*)(Xf + (size_t)(m0 + row) * 1024 + k0 + cc);
#pragma unroll
                for (int q = 0; q < 4; ++q) hv[q] = (_Float16)x[q];
            }
            *(v4h*)&As[row * LD + cc] = hv;
        }
        // --- stage B = rows of W (so B(k,n)=W[n,k] with k contiguous) ---
#pragma unroll
        for (int j = 0; j < 4; ++j) {
            const int e   = t + j * 256;
            const int row = e >> 3;
            const int cc  = (e & 7) * 4;
            v4f x = *(const v4f*)(W + (size_t)(n0 + row) * 1024 + k0 + cc);
            v4h hv;
#pragma unroll
            for (int q = 0; q < 4; ++q) hv[q] = (_Float16)x[q];
            *(v4h*)&Bs[row * LD + cc] = hv;
        }
        __syncthreads();

        // --- fragments per ISA 16-bit layouts ---
        v16h af[2];
#pragma unroll
        for (int i = 0; i < 2; ++i) {
            const int r = wm + i * 16 + ll;
            const int b = hl * 8;
            v8h lo = *(const v8h*)&As[r * LD + b];
            v8h hi = *(const v8h*)&As[r * LD + b + 16];
            af[i] = combine8(lo, hi);
        }
#pragma unroll
        for (int n = 0; n < 4; ++n) {
            const int c  = wn + n * 16 + ll;
            const int kb = hl * 8;
            v8h lo = *(const v8h*)&Bs[c * LD + kb];
            v8h hi = *(const v8h*)&Bs[c * LD + kb + 16];
            v16h bf = combine8(lo, hi);
#pragma unroll
            for (int i = 0; i < 2; ++i) acc[i][n] = wmma_f16(af[i], bf, acc[i][n]);
        }
    }

    // --- epilogue ---
#pragma unroll
    for (int i = 0; i < 2; ++i)
#pragma unroll
        for (int n = 0; n < 4; ++n)
#pragma unroll
            for (int r = 0; r < 8; ++r) {
                const int row = m0 + wm + i * 16 + r + 8 * hl;
                const int col = n0 + wn + n * 16 + ll;
                const float v = acc[i][n][r] + bias[col];
                if (mode == 0) {
                    const int b  = row / S, s = row - b * S;
                    const int h  = col >> 6, hd = col & 63;
                    outH[(((size_t)(b * 16 + h)) * S + s) * 64 + hd] = (_Float16)v;
                } else {
                    outF[(size_t)row * 1024 + col] = v;
                }
            }
}

// ---------------------------------------------------------------------------
// Flash attention: one block = 128 q-rows of one (b,h); wave owns 32 rows.
// K tile staged via async global->LDS DMA (ASYNCcnt); V staged transposed.
// ---------------------------------------------------------------------------
__global__ __launch_bounds__(128)
void flash_attn(const _Float16* __restrict__ qh, const _Float16* __restrict__ kh,
                const _Float16* __restrict__ vh, const float* __restrict__ bias,
                const int* __restrict__ amask, _Float16* __restrict__ ctx)
{
    constexpr int LDK = 72, LDV = 40, LDP = 40;
    __shared__ _Float16 Ks[32 * LDK];      // [kv][d]
    __shared__ _Float16 Vt[64 * LDV];      // [d][kv]   (transposed for B-frags)
    __shared__ _Float16 Ps[4][16 * LDP];   // per-wave P tile

    const int t    = threadIdx.x;
    const int lane = t & 31;
    const int wv   = t >> 5;
    const int ll   = lane & 15;
    const int hl   = lane >> 4;
    const int h    = blockIdx.y;
    const int bz   = blockIdx.z;
    const int q0   = blockIdx.x * 128 + wv * 32;

    const size_t headOff  = ((size_t)(bz * 16 + h)) * 2048 * 64;
    const size_t biasBase = ((size_t)h) * 2048 * 2048;

    // Q fragments live in registers for the whole kv sweep
    v16h aq[2][2];
#pragma unroll
    for (int i = 0; i < 2; ++i)
#pragma unroll
        for (int ks = 0; ks < 2; ++ks) {
            const _Float16* p = qh + headOff + (size_t)(q0 + i * 16 + ll) * 64 + hl * 8 + ks * 32;
            aq[i][ks] = combine8(*(const v8h*)p, *(const v8h*)(p + 16));
        }

    float mrun[2][8], lrun[2][8];
    v8f   cacc[2][4];
#pragma unroll
    for (int i = 0; i < 2; ++i)
#pragma unroll
        for (int r = 0; r < 8; ++r) { mrun[i][r] = -3.0e38f; lrun[i][r] = 0.f; }
#pragma unroll
    for (int i = 0; i < 2; ++i)
#pragma unroll
        for (int n = 0; n < 4; ++n) cacc[i][n] = (v8f){0.f,0.f,0.f,0.f,0.f,0.f,0.f,0.f};

    for (int kt = 0; kt < 64; ++kt) {
        const int kv0 = kt * 32;
        __syncthreads();
        {   // cooperative K/V staging: 128 threads, each 16 halfs of one row
            const int row = t >> 2;
            const int dc  = (t & 3) * 16;
            const _Float16* pk = kh + headOff + (size_t)(kv0 + row) * 64 + dc;
#if HAVE_ASYNC_LDS
            // CDNA5 async DMA: copies global memory straight into LDS, 16B per lane, no VGPR data
            __builtin_amdgcn_global_load_async_to_lds_b128(
                (as1_v4i*)pk,
                (as3_v4i*)&Ks[row * LDK + dc], 0, 0);
            __builtin_amdgcn_global_load_async_to_lds_b128(
                (as1_v4i*)(pk + 8),
                (as3_v4i*)&Ks[row * LDK + dc + 8], 0, 0);
#else
            v8h a0 = *(const v8h*)pk, a1 = *(const v8h*)(pk + 8);
            *(v8h*)&Ks[row * LDK + dc]     = a0;
            *(v8h*)&Ks[row * LDK + dc + 8] = a1;
#endif
            const _Float16* pv = vh + headOff + (size_t)(kv0 + row) * 64 + dc;
            v8h b0 = *(const v8h*)pv, b1 = *(const v8h*)(pv + 8);
#pragma unroll
            for (int jj = 0; jj < 8; ++jj) {
                Vt[(dc + jj) * LDV + row]     = b0[jj];
                Vt[(dc + 8 + jj) * LDV + row] = b1[jj];
            }
        }
#if HAVE_ASYNC_LDS
        asm volatile("s_wait_asynccnt 0x0" ::: "memory");
#endif
        __syncthreads();

        if (kt + 1 < 64)   // pull next bias tile toward L2 (global_prefetch_b8)
            __builtin_prefetch(bias + biasBase + (size_t)(q0 + lane) * 2048 + kv0 + 32, 0, 0);

        const int mk0 = amask[bz * 2048 + kv0 + ll];
        const int mk1 = amask[bz * 2048 + kv0 + 16 + ll];

#pragma unroll
        for (int i = 0; i < 2; ++i) {
            // S = Q·K^T over HD=64 (2 chained WMMAs per 16-col tile)
            v8f s[2];
#pragma unroll
            for (int j = 0; j < 2; ++j) {
                const int kr = j * 16 + ll;
                v16h b0 = combine8(*(const v8h*)&Ks[kr * LDK + hl * 8],
                                   *(const v8h*)&Ks[kr * LDK + hl * 8 + 16]);
                v16h b1 = combine8(*(const v8h*)&Ks[kr * LDK + 32 + hl * 8],
                                   *(const v8h*)&Ks[kr * LDK + 32 + hl * 8 + 16]);
                v8f z = (v8f){0.f,0.f,0.f,0.f,0.f,0.f,0.f,0.f};
                z = wmma_f16(aq[i][0], b0, z);
                s[j] = wmma_f16(aq[i][1], b1, z);
            }
            // scale + bias + mask, per-row online softmax
            float mt[8];
#pragma unroll
            for (int r = 0; r < 8; ++r) {
                const size_t brow = biasBase + (size_t)(q0 + i * 16 + r + 8 * hl) * 2048 + kv0;
                float s0 = s[0][r] * 0.125f + bias[brow + ll];
                float s1 = s[1][r] * 0.125f + bias[brow + 16 + ll];
                if (mk0 == 0) s0 = -3.0e38f;
                if (mk1 == 0) s1 = -3.0e38f;
                s[0][r] = s0; s[1][r] = s1;
                float m = fmaxf(s0, s1);
                m = fmaxf(m, __shfl_xor(m, 1, 32));
                m = fmaxf(m, __shfl_xor(m, 2, 32));
                m = fmaxf(m, __shfl_xor(m, 4, 32));
                m = fmaxf(m, __shfl_xor(m, 8, 32));
                mt[r] = m;
            }
#pragma unroll
            for (int r = 0; r < 8; ++r) {
                const float mnew = fmaxf(mrun[i][r], mt[r]);
                const float cs   = __expf(mrun[i][r] - mnew);
                mrun[i][r] = mnew;
                const float p0 = __expf(s[0][r] - mnew);
                const float p1 = __expf(s[1][r] - mnew);
                float ls = p0 + p1;
                ls += __shfl_xor(ls, 1, 32);
                ls += __shfl_xor(ls, 2, 32);
                ls += __shfl_xor(ls, 4, 32);
                ls += __shfl_xor(ls, 8, 32);
                lrun[i][r] = lrun[i][r] * cs + ls;
#pragma unroll
                for (int n = 0; n < 4; ++n) cacc[i][n][r] *= cs;
                const int prow = r + 8 * hl;
                Ps[wv][prow * LDP + ll]      = (_Float16)p0;
                Ps[wv][prow * LDP + 16 + ll] = (_Float16)p1;
            }
            asm volatile("s_wait_dscnt 0" ::: "memory");   // same-wave P re-layout via LDS
            v16h pa = combine8(*(const v8h*)&Ps[wv][ll * LDP + hl * 8],
                               *(const v8h*)&Ps[wv][ll * LDP + hl * 8 + 16]);
#pragma unroll
            for (int n = 0; n < 4; ++n) {
                const int d = n * 16 + ll;
                v16h bv = combine8(*(const v8h*)&Vt[d * LDV + hl * 8],
                                   *(const v8h*)&Vt[d * LDV + hl * 8 + 16]);
                cacc[i][n] = wmma_f16(pa, bv, cacc[i][n]);
            }
        }
    }

    // normalize + store ctx as [B,SQ,D] f16 (heads concatenated)
#pragma unroll
    for (int i = 0; i < 2; ++i)
#pragma unroll
        for (int n = 0; n < 4; ++n)
#pragma unroll
            for (int r = 0; r < 8; ++r) {
                const int row = q0 + i * 16 + r + 8 * hl;
                const int col = h * 64 + n * 16 + ll;
                const float v = cacc[i][n][r] / lrun[i][r];
                ctx[(size_t)(bz * 2048 + row) * 1024 + col] = (_Float16)v;
            }
}

// ---------------------------------------------------------------------------
// Residual + LayerNorm: one block per row
// ---------------------------------------------------------------------------
__global__ __launch_bounds__(256)
void residual_ln(const float* __restrict__ query, const float* __restrict__ oproj,
                 const float* __restrict__ gamma, const float* __restrict__ beta,
                 float* __restrict__ out)
{
    __shared__ float red[2][8];
    const int row  = blockIdx.x;
    const int t    = threadIdx.x;
    const int lane = t & 31;
    const int wv   = t >> 5;
    const size_t base = (size_t)row * 1024;

    float x[4], s = 0.f, s2 = 0.f;
#pragma unroll
    for (int j = 0; j < 4; ++j) {
        const int c = t + j * 256;
        const float v = query[base + c] + oproj[base + c];
        x[j] = v; s += v; s2 += v * v;
    }
#pragma unroll
    for (int off = 1; off < 32; off <<= 1) {
        s  += __shfl_xor(s, off, 32);
        s2 += __shfl_xor(s2, off, 32);
    }
    if (lane == 0) { red[0][wv] = s; red[1][wv] = s2; }
    __syncthreads();
    if (wv == 0) {
        float a  = (lane < 8) ? red[0][lane] : 0.f;
        float b2 = (lane < 8) ? red[1][lane] : 0.f;
#pragma unroll
        for (int off = 1; off < 8; off <<= 1) {
            a  += __shfl_xor(a, off, 32);
            b2 += __shfl_xor(b2, off, 32);
        }
        if (lane == 0) { red[0][0] = a; red[1][0] = b2; }
    }
    __syncthreads();
    const float mean = red[0][0] * (1.0f / 1024.0f);
    const float var  = red[1][0] * (1.0f / 1024.0f) - mean * mean;
    const float inv  = rsqrtf(var + 1e-5f);
#pragma unroll
    for (int j = 0; j < 4; ++j) {
        const int c = t + j * 256;
        out[base + c] = (x[j] - mean) * inv * gamma[c] + beta[c];
    }
}

// ---------------------------------------------------------------------------
extern "C" void kernel_launch(void* const* d_in, const int* in_sizes, int n_in,
                              void* d_out, int out_size, void* d_ws, size_t ws_size,
                              hipStream_t stream) {
    (void)in_sizes; (void)n_in; (void)out_size; (void)ws_size;
    const float* query = (const float*)d_in[0];
    const float* key   = (const float*)d_in[1];
    const float* value = (const float*)d_in[2];
    const int*   amask = (const int*)d_in[3];
    const float* rpb   = (const float*)d_in[4];
    const float* Wq = (const float*)d_in[5];
    const float* bq = (const float*)d_in[6];
    const float* Wk = (const float*)d_in[7];
    const float* bk = (const float*)d_in[8];
    const float* Wv = (const float*)d_in[9];
    const float* bv = (const float*)d_in[10];
    const float* Wo = (const float*)d_in[11];
    const float* bo = (const float*)d_in[12];
    const float* ga = (const float*)d_in[13];
    const float* be = (const float*)d_in[14];
    float* out = (float*)d_out;

    char* ws = (char*)d_ws;
    const size_t MB16 = 16ull << 20;
    _Float16* qh  = (_Float16*)(ws);
    _Float16* kh  = (_Float16*)(ws + MB16);
    _Float16* vh  = (_Float16*)(ws + 2 * MB16);
    _Float16* ctx = (_Float16*)(ws + 3 * MB16);
    float*    opj = (float*)(ws + MB16);   // aliases kh/vh; only used after attention

    const dim3 gg(1024 / 128, 8192 / 128);
    proj_gemm<false><<<gg, 256, 0, stream>>>(query, nullptr, Wq, bq, qh, nullptr, 8192, 2048, 0);
    proj_gemm<false><<<gg, 256, 0, stream>>>(key,   nullptr, Wk, bk, kh, nullptr, 8192, 2048, 0);
    proj_gemm<false><<<gg, 256, 0, stream>>>(value, nullptr, Wv, bv, vh, nullptr, 8192, 2048, 0);
    flash_attn<<<dim3(16, 16, 4), 128, 0, stream>>>(qh, kh, vh, rpb, amask, ctx);
    proj_gemm<true><<<gg, 256, 0, stream>>>(nullptr, ctx, Wo, bo, nullptr, opj, 8192, 2048, 1);
    residual_ln<<<8192, 256, 0, stream>>>(query, opj, ga, be, out);
}
